// NCM_30468497998426
// MI455X (gfx1250) — matrix operand order repaired and checked by reading, not compiled, not run on hardware
//
#include <hip/hip_runtime.h>
#include <hip/hip_bf16.h>

// ---------------------------------------------------------------------------
// Few-shot nearest-class classifier. C=1000 classes, S=5 shots, D=512, Q=5000.
// argmin-euclid == argmax-cosine for L2-normalized vectors, so one path covers
// both use_cosine branches.
//
//   1) mean over support vectors
//   2) center + L2-normalize + bf16 convert (padded: support 5040, query 5024)
//   3) per block: TDM-stage 80x512 A tile + 32x512 B tile into LDS
//      (tensor_load_to_lds + s_wait_tensorcnt), then 10 waves of
//      v_wmma_f32_16x16x32_bf16 from ds_load_b128 fragments; shot-max +
//      class argmax; global argmax via atomicMax on packed u64 keys
//   4) unpack keys -> int32 class ids
// ---------------------------------------------------------------------------

#define D 512
#define NSUP 5000      // C*S
#define NSUP_PAD 5040  // 63 * 80
#define NQ 5000
#define NQ_PAD 5024    // 157 * 32
#define MB 80          // support rows per block tile (= 16 classes exactly)
#define NB 32          // queries per block tile

#define LDS_A_OFF 0
#define LDS_B_OFF (MB * D * 2)              //  81920
#define LDS_T_OFF (LDS_B_OFF + NB * D * 2)  // 114688
#define LDS_TOTAL (LDS_T_OFF + MB * NB * 4) // 124928 (< 320KB/WGP)

typedef __attribute__((ext_vector_type(16))) __bf16 bf16x16;
typedef __attribute__((ext_vector_type(8)))  __bf16 bf16x8;
typedef __attribute__((ext_vector_type(8)))  float  f32x8;
typedef __attribute__((ext_vector_type(4)))  unsigned int u32x4;
typedef __attribute__((ext_vector_type(4)))  int i32x4;
typedef __attribute__((ext_vector_type(8)))  int i32x8;

__device__ __forceinline__ unsigned short f2bf(float f) {
    unsigned int u = __float_as_uint(f);
    unsigned int r = u + 0x7FFFu + ((u >> 16) & 1u);   // round-to-nearest-even
    return (unsigned short)(r >> 16);
}

// ---- pass 1: mean over the 5000 support vectors ---------------------------
__global__ void k_mean(const float* __restrict__ sup, float* __restrict__ mean) {
    int d = blockIdx.x * blockDim.x + threadIdx.x;
    if (d >= D) return;
    float acc = 0.0f;
    for (int r = 0; r < NSUP; ++r) acc += sup[r * D + d];   // coalesced
    mean[d] = acc * (1.0f / (float)NSUP);
}

// ---- pass 2: center + normalize + bf16 convert (one block per row) --------
__global__ __launch_bounds__(128)
void k_norm(const float* __restrict__ x, const float* __restrict__ mean,
            unsigned short* __restrict__ out, int nreal) {
    const int row = blockIdx.x;
    const int t   = threadIdx.x;               // 128 threads, 4 elems each
    if (row >= nreal) {                        // zero padding rows
        ushort4 z = {0, 0, 0, 0};
        ((ushort4*)(out + (size_t)row * D))[t] = z;
        return;
    }
    __shared__ float red[128];
    float4 xin = ((const float4*)(x + (size_t)row * D))[t];
    float4 m4  = ((const float4*)mean)[t];
    float v0 = xin.x - m4.x, v1 = xin.y - m4.y, v2 = xin.z - m4.z, v3 = xin.w - m4.w;
    red[t] = v0 * v0 + v1 * v1 + v2 * v2 + v3 * v3;
    __syncthreads();
    for (int off = 64; off > 0; off >>= 1) {
        if (t < off) red[t] += red[t + off];
        __syncthreads();
    }
    float scale = rsqrtf(red[0]);
    ushort4 o;
    o.x = f2bf(v0 * scale); o.y = f2bf(v1 * scale);
    o.z = f2bf(v2 * scale); o.w = f2bf(v3 * scale);
    ((ushort4*)(out + (size_t)row * D))[t] = o;
}

// ---- pass 3a: reset argmax keys ------------------------------------------
__global__ void k_init_keys(unsigned long long* __restrict__ keys, int n) {
    int i = blockIdx.x * blockDim.x + threadIdx.x;
    if (i < n) keys[i] = 0ull;   // all real keys are > 0 (sims >= -1)
}

#if __has_builtin(__builtin_amdgcn_tensor_load_to_lds)
// Issue one TDM 2-D tile load: rows x 512 bf16, row-major, into LDS.
// D# layout per CDNA5 ISA ch.8 (group0: count/lds/global/type; group1:
// data_size, tensor dims, tile dims, strides).
__device__ __forceinline__ void tdm_load_tile(unsigned lds_addr, const void* gptr,
                                              unsigned rows, unsigned tensor_rows) {
    unsigned long long ga = (unsigned long long)(size_t)gptr;
    u32x4 g0;
    g0[0] = 1u;                                           // count=1 (user mode)
    g0[1] = lds_addr;                                     // LDS byte address
    g0[2] = (unsigned)(ga & 0xFFFFFFFFull);               // global_addr[31:0]
    g0[3] = (unsigned)((ga >> 32) & 0x01FFFFFFull)        // global_addr[56:32]
            | (2u << 30);                                 // type=2 ("image")
    i32x8 g1;
    g1[0] = (int)(1u << 16);                              // data_size=1 (2 bytes)
    g1[1] = (int)(512u << 16);                            // tensor_dim0 lo16
    g1[2] = (int)(tensor_rows << 16);                     // dim0 hi16=0 | tensor_dim1 lo16
    g1[3] = (int)((tensor_rows >> 16) | (512u << 16));    // tensor_dim1 hi | tile_dim0=512
    g1[4] = (int)(rows & 0xFFFFu);                        // tile_dim1=rows, tile_dim2=0
    g1[5] = 512;                                          // tensor_dim0_stride lo32
    g1[6] = 0;                                            // stride hi | dim1_stride lo
    g1[7] = 0;
    i32x4 z4 = {0, 0, 0, 0};
#if __clang_major__ >= 23
    i32x8 z8 = {0, 0, 0, 0, 0, 0, 0, 0};
    __builtin_amdgcn_tensor_load_to_lds(g0, g1, z4, z4, z8, 0);
#else
    __builtin_amdgcn_tensor_load_to_lds(g0, g1, z4, z4, 0);
#endif
}
#endif

// ---- pass 3b: WMMA similarity GEMM + shot-max + class argmax --------------
// Block: 320 threads = 10 waves; 5 M-tiles x 2 N-tiles; A/B tiles in LDS.
__global__ __launch_bounds__(320)
void k_sim(const unsigned short* __restrict__ Sn,  // [NSUP_PAD][D] bf16
           const unsigned short* __restrict__ Qn,  // [NQ_PAD][D]   bf16
           unsigned long long* __restrict__ keys) {
    extern __shared__ char smem[];

    const int tid   = threadIdx.x;
    const int lane  = tid & 31;
    const int wid   = tid >> 5;           // 0..9
    const int mt    = wid % 5;
    const int nt    = wid / 5;
    const int mbase = blockIdx.y * MB;
    const int nbase = blockIdx.x * NB;
    const int r     = lane & 15;
    const int hi    = lane >> 4;

    // ---- stage A (80x512) and B (32x512) bf16 tiles into LDS --------------
#if __has_builtin(__builtin_amdgcn_tensor_load_to_lds)
    if (tid < 32) {                        // wave 0 issues both TDM descriptors
        unsigned ldsA = (unsigned)(size_t)(smem + LDS_A_OFF);
        unsigned ldsB = (unsigned)(size_t)(smem + LDS_B_OFF);
        tdm_load_tile(ldsA, Sn + (size_t)mbase * D, MB, NSUP_PAD);
        tdm_load_tile(ldsB, Qn + (size_t)nbase * D, NB, NQ_PAD);
        __builtin_amdgcn_s_wait_tensorcnt(0);   // TENSORcnt == 0
    }
#else
    {   // cooperative 16B-chunk copy fallback
        const uint4* gA = (const uint4*)(Sn + (size_t)mbase * D);
        const uint4* gB = (const uint4*)(Qn + (size_t)nbase * D);
        uint4* sA = (uint4*)(smem + LDS_A_OFF);
        uint4* sB = (uint4*)(smem + LDS_B_OFF);
        for (int i = tid; i < MB * D / 8; i += 320) sA[i] = gA[i];
        for (int i = tid; i < NB * D / 8; i += 320) sB[i] = gB[i];
    }
#endif
    __syncthreads();

    // ---- WMMA over K=512 from LDS fragments (ds_load_b128) ----------------
    // A (16x32 bf16): lane half 0 -> K 0..7 & 16..23 ; half 1 -> K 8..15 & 24..31
    // B (32x16 bf16): lane half 0 -> K 0..7 & 16..23 ; half 1 -> K 8..15 & 24..31 (col = lane%16)
    const __bf16* aL = (const __bf16*)(smem + LDS_A_OFF) + (mt * 16 + r) * D;
    const __bf16* bL = (const __bf16*)(smem + LDS_B_OFF) + (nt * 16 + r) * D;
    const int aoff = hi * 8;
    const int boff = hi * 8;

    f32x8 c = {};
    for (int kk = 0; kk < D; kk += 32) {
        union { bf16x16 v; bf16x8 h[2]; } A, B;
        A.h[0] = *(const bf16x8*)(aL + kk + aoff);
        A.h[1] = *(const bf16x8*)(aL + kk + aoff + 16);
        B.h[0] = *(const bf16x8*)(bL + kk + boff);
        B.h[1] = *(const bf16x8*)(bL + kk + boff + 16);
        c = __builtin_amdgcn_wmma_f32_16x16x32_bf16(
                false, A.v, false, B.v, (short)0, c, false, false);
    }
    __syncthreads();   // A/B LDS no longer needed; reuse region for results? (tile is separate)

    // ---- C/D tile -> LDS ---------------------------------------------------
    float* tile = (float*)(smem + LDS_T_OFF);
    #pragma unroll
    for (int i = 0; i < 8; ++i) {
        int ml = mt * 16 + hi * 8 + i;
        int nl = nt * 16 + r;
        tile[ml * NB + nl] = c[i];
    }
    __syncthreads();

    // ---- shot-max + per-block class argmax + global atomic argmax ---------
    if (tid < NB) {
        int q = nbase + tid;
        if (q < NQ) {
            const int cbase = blockIdx.y * 16;
            float best = -3.0e38f;
            int   bc   = 0;
            for (int cc = 0; cc < 16; ++cc) {
                int cg = cbase + cc;
                if (cg >= 1000) break;                 // padded classes
                float m = tile[(cc * 5 + 0) * NB + tid];
                #pragma unroll
                for (int s = 1; s < 5; ++s)
                    m = fmaxf(m, tile[(cc * 5 + s) * NB + tid]);
                if (m > best) { best = m; bc = cg; }   // strict > : lowest class wins
            }
            unsigned int u = __float_as_uint(best);    // order-preserving map
            u = (u & 0x80000000u) ? ~u : (u | 0x80000000u);
            unsigned long long key =
                ((unsigned long long)u << 32) |
                (unsigned long long)(0xFFFFFFFFu - (unsigned)bc);
            atomicMax(keys + q, key);                  // global_atomic_max_u64
        }
    }
}

// ---- pass 4: unpack keys -> int32 class ids -------------------------------
__global__ void k_extract(const unsigned long long* __restrict__ keys,
                          int* __restrict__ out) {
    int q = blockIdx.x * blockDim.x + threadIdx.x;
    if (q < NQ)
        out[q] = (int)(0xFFFFFFFFu - (unsigned int)(keys[q] & 0xFFFFFFFFull));
}

extern "C" void kernel_launch(void* const* d_in, const int* in_sizes, int n_in,
                              void* d_out, int out_size, void* d_ws, size_t ws_size,
                              hipStream_t stream) {
    const float* sup = (const float*)d_in[0];   // [1000,5,512]
    const float* qry = (const float*)d_in[1];   // [5000,512]
    // d_in[2] = use_cosine: both branches give identical predictions for
    // normalized vectors (euclid = sqrt(2-2*cos) is monotone in cos).
    (void)in_sizes; (void)n_in; (void)out_size; (void)ws_size;

    char* ws = (char*)d_ws;
    float*              mean = (float*)(ws);                        //   2 KB
    unsigned long long* keys = (unsigned long long*)(ws + 2048);    //  40 KB
    unsigned short*     Sn   = (unsigned short*)(ws + 43008);       // 5.16 MB
    unsigned short*     Qn   = (unsigned short*)(ws + 43008 + (size_t)NSUP_PAD * D * 2);
    int*                out  = (int*)d_out;

    k_mean<<<dim3((D + 255) / 256), 256, 0, stream>>>(sup, mean);
    k_norm<<<dim3(NSUP_PAD), 128, 0, stream>>>(sup, mean, Sn, NSUP);
    k_norm<<<dim3(NQ_PAD),   128, 0, stream>>>(qry, mean, Qn, NQ);
    k_init_keys<<<dim3((NQ_PAD + 255) / 256), 256, 0, stream>>>(keys, NQ_PAD);
    k_sim<<<dim3(NQ_PAD / NB, NSUP_PAD / MB), 320, LDS_TOTAL, stream>>>(Sn, Qn, keys);
    k_extract<<<dim3((NQ + 255) / 256), 256, 0, stream>>>(keys, out);
}
